// HiResPrecipNet_subpixel_89223650607424
// MI455X (gfx1250) — compile-verified, matrix operand order checked by reading, not run on hardware
//
#include <hip/hip_runtime.h>
#include <hip/hip_bf16.h>

// ---------------------------------------------------------------------------
// HiResPrecipNet for MI455X (gfx1250, wave32).
// Dense transforms run on v_wmma_f32_16x16x32_f16 (f16 in / f32 accumulate),
// with the B panel staged into LDS by the Tensor Data Mover (tensor_load_to_lds
// + s_wait_tensorcnt). Graph edge ops use global f32 atomics; BatchNorm apply
// fuses the f16 cast.
// ---------------------------------------------------------------------------

typedef _Float16 half16 __attribute__((ext_vector_type(16)));
typedef float    float8 __attribute__((ext_vector_type(8)));
typedef unsigned int uint4v __attribute__((ext_vector_type(4)));
typedef int          int4v  __attribute__((ext_vector_type(4)));
typedef int          int8v  __attribute__((ext_vector_type(8)));

#define TPB 256

static inline int cdivll(long long a, long long b) { return (int)((a + b - 1) / b); }

// ------------------------------ CNN -----------------------------------------

__global__ void conv2d_tanh(const float* __restrict__ x, const float* __restrict__ w,
                            const float* __restrict__ bias, float* __restrict__ y,
                            int nimg, int cin, int cout, int ksz, int pad)
{
    long long idx = (long long)blockIdx.x * blockDim.x + threadIdx.x;
    long long total = (long long)nimg * cout * 32 * 32;
    if (idx >= total) return;
    int xx = (int)(idx & 31);
    int yy = (int)((idx >> 5) & 31);
    int co = (int)((idx >> 10) % cout);
    int n  = (int)(idx / ((long long)1024 * cout));
    float acc = bias[co];
    for (int ci = 0; ci < cin; ++ci) {
        const float* xp = x + (((size_t)n * cin + ci) * 32) * 32;
        const float* wp = w + (((size_t)co * cin + ci) * ksz) * ksz;
        for (int ky = 0; ky < ksz; ++ky) {
            int sy = yy + ky - pad;
            if ((unsigned)sy >= 32u) continue;
            for (int kx = 0; kx < ksz; ++kx) {
                int sx = xx + kx - pad;
                if ((unsigned)sx >= 32u) continue;
                acc += xp[sy * 32 + sx] * wp[ky * ksz + kx];
            }
        }
    }
    y[idx] = tanhf(acc);
}

// conv3 (32->45, 3x3) fused with pixel_shuffle(r=3), sigmoid, and the
// [b,5,5,c,H,W]->[b*H*W, 125] permute, written directly as f16 padded to 128.
__global__ void conv3_shuffle_sig(const float* __restrict__ x, const float* __restrict__ w,
                                  const float* __restrict__ bias, _Float16* __restrict__ hbuf,
                                  int nimg)
{
    long long idx = (long long)blockIdx.x * blockDim.x + threadIdx.x;
    long long total = (long long)nimg * 45 * 32 * 32;
    if (idx >= total) return;
    int xx = (int)(idx & 31);
    int yy = (int)((idx >> 5) & 31);
    int co = (int)((idx >> 10) % 45);
    int n  = (int)(idx / (1024LL * 45));
    float acc = bias[co];
    for (int ci = 0; ci < 32; ++ci) {
        const float* xp = x + (((size_t)n * 32 + ci) * 32) * 32;
        const float* wp = w + (((size_t)co * 32 + ci) * 3) * 3;
        for (int ky = 0; ky < 3; ++ky) {
            int sy = yy + ky - 1;
            if ((unsigned)sy >= 32u) continue;
            for (int kx = 0; kx < 3; ++kx) {
                int sx = xx + kx - 1;
                if ((unsigned)sx >= 32u) continue;
                acc += xp[sy * 32 + sx] * wp[ky * 3 + kx];
            }
        }
    }
    float v = 1.0f / (1.0f + expf(-acc));
    int g = n % 25, bimg = n / 25;
    int c5 = co / 9, ry = (co % 9) / 3, rx = co % 3;
    int hy = yy * 3 + ry, hx = xx * 3 + rx;
    long long node = (long long)bimg * 9216 + hy * 96 + hx;
    hbuf[node * 128 + g * 5 + c5] = (_Float16)v;
}

// ---------------------- TDM: stage B panel into LDS -------------------------
// 2D descriptor: tile == whole [Kpad x Npad] f16 panel, contiguous in memory.
__device__ static inline void tdm_load_b(const _Float16* B, unsigned ldsoff,
                                         int Kpad, int Npad)
{
    unsigned long long ga = (unsigned long long)(size_t)B;
    uint4v g0 = (uint4v)0u;
    g0[0] = 1u;                                                  // count = 1
    g0[1] = ldsoff;                                              // lds_addr (bytes)
    g0[2] = (unsigned)ga;                                        // global_addr[31:0]
    g0[3] = (unsigned)((ga >> 32) & 0x01FFFFFFull) | (2u << 30); // addr[56:32] | type=2
    int8v g1 = (int8v)0;
    g1[0] = 1 << 16;                 // workgroup_mask=0 (not in cluster), data_size=1 (2B)
    g1[1] = (Npad & 0xFFFF) << 16;   // tensor_dim0[15:0]  (elements)
    g1[2] = (Kpad & 0xFFFF) << 16;   // tensor_dim1[15:0]
    g1[3] = (Npad & 0xFFFF) << 16;   // tile_dim0
    g1[4] = (Kpad & 0xFFFF);         // tile_dim1 (tile_dim2 = 0)
    g1[5] = Npad;                    // tensor_dim0_stride[31:0]
    int4v gz = (int4v)0;
#if defined(__clang_major__) && (__clang_major__ >= 23)
    int8v gz8 = (int8v)0;
    __builtin_amdgcn_tensor_load_to_lds(g0, g1, gz, gz, gz8, 0);
#else
    __builtin_amdgcn_tensor_load_to_lds(g0, g1, gz, gz, 0);
#endif
}

// --------------------------- WMMA GEMM --------------------------------------
// C[M,ldc] = act( A_f16[M,KPAD] * B_f16[KPAD,Npad] + bias ), one 16x16 tile/wave.
// B panel staged to LDS via the Tensor Data Mover; K-loop fully unrolled.
template <int KPAD>
__global__ void gemm_wmma(const _Float16* __restrict__ A,
                          const _Float16* __restrict__ B, int Npad,
                          const float* __restrict__ bias, int nbias,
                          float* __restrict__ C, int ldc, int nstore,
                          int M, int act)
{
    __shared__ _Float16 Bs[128 * 128];           // 32 KB (>= any B panel here)

    if (threadIdx.x < 32) {                      // wave 0 drives the TDM
        tdm_load_b(B, (unsigned)(size_t)&Bs[0], KPAD, Npad);
        __builtin_amdgcn_s_wait_tensorcnt(0);
    }
    __syncthreads();

    int ntiles = Npad >> 4;
    int mtiles = M >> 4;
    int wave = blockIdx.x * (blockDim.x >> 5) + (threadIdx.x >> 5);
    if (wave >= ntiles * mtiles) return;         // wave-uniform: EXEC stays all-1s
    int tm = wave / ntiles, tn = wave % ntiles;
    int lane = threadIdx.x & 31;
    int row  = lane & 15;
    int half = lane >> 4;
    int m0 = tm << 4, n0 = tn << 4;

    float8 acc = {};
#pragma unroll
    for (int kb = 0; kb < KPAD; kb += 32) {
        const _Float16* Ap = A + (size_t)(m0 + row) * KPAD + kb;
        __builtin_prefetch(Ap + 32, 0, 0);       // global_prefetch_b8 on next K block
        half16 a, b;
#pragma unroll
        for (int v = 0; v < 8; ++v) {
            int kk = (v < 4) ? (half * 8 + 2 * v) : (16 + half * 8 + 2 * (v - 4));
            a[2 * v]     = Ap[kk];
            a[2 * v + 1] = Ap[kk + 1];
            b[2 * v]     = Bs[(kb + kk) * Npad + n0 + row];
            b[2 * v + 1] = Bs[(kb + kk + 1) * Npad + n0 + row];
        }
        acc = __builtin_amdgcn_wmma_f32_16x16x32_f16(false, a, false, b,
                                                     (short)0, acc, false, false);
    }
#pragma unroll
    for (int i = 0; i < 8; ++i) {
        int r = m0 + i + 8 * half;
        int c = n0 + row;
        if (c < nstore && r < M) {
            float v = acc[i] + ((c < nbias) ? bias[c] : 0.0f);
            if (act == 1) v = fmaxf(v, 0.0f);
            C[(size_t)r * ldc + c] = v;
        }
    }
}

static void launch_gemm(int Kpad, const _Float16* A, const _Float16* B, int Npad,
                        const float* bias, int nbias, float* C, int ldc, int nstore,
                        int M, int act, hipStream_t stream)
{
    int tiles = (M >> 4) * (Npad >> 4);
    int blocks = cdivll(tiles, 8);
    switch (Kpad) {
        case 32:
            gemm_wmma<32><<<blocks, TPB, 0, stream>>>(A, B, Npad, bias, nbias, C, ldc, nstore, M, act);
            break;
        case 64:
            gemm_wmma<64><<<blocks, TPB, 0, stream>>>(A, B, Npad, bias, nbias, C, ldc, nstore, M, act);
            break;
        default:
            gemm_wmma<128><<<blocks, TPB, 0, stream>>>(A, B, Npad, bias, nbias, C, ldc, nstore, M, act);
            break;
    }
}

// W[N,K] fp32 row-major  ->  B[Kpad,Npad] f16 (W^T, zero padded)
__global__ void pack_weight_f16(const float* __restrict__ W, _Float16* __restrict__ Bp,
                                int K, int N, int Kpad, int Npad)
{
    int idx = blockIdx.x * blockDim.x + threadIdx.x;
    if (idx >= Kpad * Npad) return;
    int n = idx % Npad, k = idx / Npad;
    Bp[idx] = (k < K && n < N) ? (_Float16)W[(size_t)n * K + k] : (_Float16)0.0f;
}

__global__ void cast_f16_pad(const float* __restrict__ X, int ldx, int D,
                             _Float16* __restrict__ Y, int Dpad, int N)
{
    long long idx = (long long)blockIdx.x * blockDim.x + threadIdx.x;
    if (idx >= (long long)N * Dpad) return;
    int c = (int)(idx % Dpad);
    long long r = idx / Dpad;
    Y[idx] = (c < D) ? (_Float16)X[r * ldx + c] : (_Float16)0.0f;
}

// ----------------------------- BatchNorm ------------------------------------

__global__ void bn_stats(const float* __restrict__ x, int ld, int N,
                         const float* __restrict__ g, const float* __restrict__ b,
                         float* __restrict__ scale, float* __restrict__ shift)
{
    int d = blockIdx.x;
    __shared__ float ssum[TPB], ssq[TPB];
    float s = 0.0f, q = 0.0f;
    for (int i = threadIdx.x; i < N; i += blockDim.x) {
        float v = x[(size_t)i * ld + d];
        s += v; q += v * v;
    }
    ssum[threadIdx.x] = s; ssq[threadIdx.x] = q;
    __syncthreads();
    for (int st = TPB / 2; st > 0; st >>= 1) {
        if (threadIdx.x < st) {
            ssum[threadIdx.x] += ssum[threadIdx.x + st];
            ssq[threadIdx.x]  += ssq[threadIdx.x + st];
        }
        __syncthreads();
    }
    if (threadIdx.x == 0) {
        float mu = ssum[0] / N;
        float var = ssq[0] / N - mu * mu;
        float sc = g[d] * rsqrtf(var + 1e-5f);
        scale[d] = sc;
        shift[d] = b[d] - mu * sc;
    }
}

__global__ void bn_apply_f16(const float* __restrict__ X, int ldx, int D,
                             const float* __restrict__ scale, const float* __restrict__ shift,
                             _Float16* __restrict__ Y, int Dpad, int N, int relu)
{
    long long idx = (long long)blockIdx.x * blockDim.x + threadIdx.x;
    if (idx >= (long long)N * Dpad) return;
    int c = (int)(idx % Dpad);
    long long r = idx / Dpad;
    float v = 0.0f;
    if (c < D) {
        v = X[r * ldx + c] * scale[c] + shift[c];
        if (relu) v = fmaxf(v, 0.0f);
    }
    Y[idx] = (_Float16)v;
}

// ------------------------------ GATv2 ---------------------------------------

__device__ inline unsigned fkey(float f) {
    unsigned u = __float_as_uint(f);
    return (u & 0x80000000u) ? ~u : (u | 0x80000000u);
}
__device__ inline float kinv(unsigned k) {
    unsigned u = (k & 0x80000000u) ? (k ^ 0x80000000u) : ~k;
    return __uint_as_float(u);
}

__global__ void gat_logits(const float* __restrict__ xl, int ldl,
                           const float* __restrict__ xr, int ldr,
                           const float* __restrict__ att, int H, int C,
                           const int* __restrict__ src, const int* __restrict__ dst,
                           int E, int nself,
                           float* __restrict__ logits, unsigned* __restrict__ mkey)
{
    long long idx = (long long)blockIdx.x * blockDim.x + threadIdx.x;
    long long tot = (long long)(E + nself) * H;
    if (idx >= tot) return;
    int e = (int)(idx / H), h = (int)(idx % H);
    int s = (e < E) ? src[e] : (e - E);
    int d = (e < E) ? dst[e] : (e - E);
    const float* pl = xl + (size_t)s * ldl + h * C;
    const float* pr = xr + (size_t)d * ldr + h * C;
    const float* pa = att + h * C;
    float acc = 0.0f;
    for (int c = 0; c < C; ++c) {
        float v = pl[c] + pr[c];
        v = (v > 0.0f) ? v : 0.2f * v;          // leaky_relu(0.2)
        acc += pa[c] * v;
    }
    logits[idx] = acc;
    atomicMax(&mkey[(size_t)d * H + h], fkey(acc));
}

__global__ void gat_exp(const float* __restrict__ logits,
                        const int* __restrict__ dst, int E, int nself, int H,
                        const unsigned* __restrict__ mkey,
                        float* __restrict__ ex, float* __restrict__ den,
                        float* __restrict__ cnt)
{
    long long idx = (long long)blockIdx.x * blockDim.x + threadIdx.x;
    long long tot = (long long)(E + nself) * H;
    if (idx >= tot) return;
    int e = (int)(idx / H), h = (int)(idx % H);
    int d = (e < E) ? dst[e] : (e - E);
    unsigned k = mkey[(size_t)d * H + h];
    float m = (k == 0u) ? 0.0f : kinv(k);       // finite-guard on empty segments
    float v = expf(logits[idx] - m);
    ex[idx] = v;
    atomicAdd(&den[(size_t)d * H + h], v);
    if (h == 0) atomicAdd(&cnt[d], 1.0f);
}

__global__ void gat_scatter(const float* __restrict__ xl, int ldl,
                            const float* __restrict__ ex, const float* __restrict__ den,
                            const int* __restrict__ src, const int* __restrict__ dst,
                            int E, int nself, int H, int C, float* __restrict__ sacc)
{
    long long idx = (long long)blockIdx.x * blockDim.x + threadIdx.x;
    int HC = H * C;
    long long tot = (long long)(E + nself) * HC;
    if (idx >= tot) return;
    int e = (int)(idx / HC), j = (int)(idx % HC), h = j / C;
    int s = (e < E) ? src[e] : (e - E);
    int d = (e < E) ? dst[e] : (e - E);
    float alpha = ex[(size_t)e * H + h] / (den[(size_t)d * H + h] + 1e-16f);
    atomicAdd(&sacc[(size_t)d * HC + j], alpha * xl[(size_t)s * ldl + j]);
}

__global__ void gat_finalize(const float* __restrict__ sacc, const float* __restrict__ cnt,
                             const float* __restrict__ bias, int HC,
                             float* __restrict__ out, int ldo, int col0, int N, int act)
{
    long long idx = (long long)blockIdx.x * blockDim.x + threadIdx.x;
    if (idx >= (long long)N * HC) return;
    int j = (int)(idx % HC);
    long long n = idx / HC;
    float v = sacc[idx] / fmaxf(cnt[n], 1.0f) + bias[j];
    if (act) v = fmaxf(v, 0.0f);
    out[n * ldo + col0 + j] = v;
}

// down-GAT xr: high_x [N,1] @ Wr[25,1]^T + br   (K=1, trivial)
__global__ void xr_down(const float* __restrict__ hx, const float* __restrict__ Wr,
                        const float* __restrict__ br, float* __restrict__ xr, int N, int C)
{
    long long idx = (long long)blockIdx.x * blockDim.x + threadIdx.x;
    if (idx >= (long long)N * C) return;
    int c = (int)(idx % C);
    long long n = idx / C;
    xr[idx] = hx[n] * Wr[c] + br[c];
}

__global__ void copy_col0(const float* __restrict__ z, float* __restrict__ x, int ld, int N)
{
    int n = blockIdx.x * blockDim.x + threadIdx.x;
    if (n < N) x[(size_t)n * ld] = z[n];
}

// ---------------------------------------------------------------------------

extern "C" void kernel_launch(void* const* d_in, const int* in_sizes, int n_in,
                              void* d_out, int out_size, void* d_ws, size_t ws_size,
                              hipStream_t stream)
{
    (void)in_sizes; (void)n_in; (void)out_size; (void)ws_size;
    const int NH = 40000, E12 = 360000, EHH = 320000, NLOW = 73728;

    // -------- inputs (JAX pytree order: dict keys sorted) --------
    const float* x_low  = (const float*)d_in[0];
    const float* high_x = (const float*)d_in[1];
    const float* z_std  = (const float*)d_in[2];
    const int* src_l2h  = (const int*)d_in[3];
    const int* dst_l2h  = (const int*)d_in[4];
    const int* src_hh   = (const int*)d_in[5];
    const int* dst_hh   = (const int*)d_in[6];
    // params.bns: [ {b,g} x 5 ]  -> idx 7..16
    const float *bn_b[5], *bn_g[5];
    for (int i = 0; i < 5; ++i) { bn_b[i] = (const float*)d_in[7 + 2*i]; bn_g[i] = (const float*)d_in[8 + 2*i]; }
    // params.cnn: b1,b2,b3,w1,w2,w3 -> 17..22
    const float* cb1 = (const float*)d_in[17];
    const float* cb2 = (const float*)d_in[18];
    const float* cb3 = (const float*)d_in[19];
    const float* cw1 = (const float*)d_in[20];
    const float* cw2 = (const float*)d_in[21];
    const float* cw3 = (const float*)d_in[22];
    // params.down: Wl,Wr,att,bias,bl,br -> 23..28
    const float* dWl = (const float*)d_in[23];
    const float* dWr = (const float*)d_in[24];
    const float* datt  = (const float*)d_in[25];
    const float* dbias = (const float*)d_in[26];
    const float* dbl = (const float*)d_in[27];
    const float* dbr = (const float*)d_in[28];
    // params.gats: 5 x {Wl,Wr,att,bias,bl,br} -> 29 + 6*l
    // params.pred: W1,W2,W3,b1,b2,b3 -> 59..64
    const float* pW1 = (const float*)d_in[59];
    const float* pW2 = (const float*)d_in[60];
    const float* pW3 = (const float*)d_in[61];
    const float* pb1 = (const float*)d_in[62];
    const float* pb2 = (const float*)d_in[63];
    const float* pb3 = (const float*)d_in[64];

    // -------- workspace arena (phase-aliased, ~125 MB) --------
    char* ws = (char*)d_ws;
    float*     bufA   = (float*)(ws);                       // 61,440,000 B (conv1out | xl+xr+s)
    float*     bufB   = (float*)(ws +  61440000);           // 30,720,000 B (conv2out | x + x_f16)
    _Float16*  hbuf   = (_Float16*)(ws + 92160000);         // 18,874,368 B (h f16 | edge buffers)
    float*     xl_low = (float*)(ws + 111034368);           //  9,437,184 B
    float*     xr_high= (float*)(ws + 120471552);           //  4,000,256 B
    _Float16*  wpack  = (_Float16*)(ws + 124471808);        //     32,768 B
    float*     bnscale= (float*)(ws + 124504576);           //        512 B
    float*     bnshift= (float*)(ws + 124505088);           //        512 B
    // edge buffers inside hbuf region (free after xl_low GEMM, stream-ordered)
    float*     logits = (float*)(ws + 92160000);            // 2,880,000 B (E*H max)
    float*     exbuf  = (float*)(ws + 95040000);            // 2,880,000 B
    unsigned*  mkey   = (unsigned*)(ws + 97920000);         //   320,000 B
    float*     den    = (float*)(ws + 98240000);            //   320,000 B
    float*     cnt    = (float*)(ws + 98560000);            //   160,000 B
    // bufA sub: xl | xr | s   (each 40000*128*4 = 20,480,000 B)
    float*     xlN  = bufA;
    float*     xrN  = (float*)((char*)bufA + 20480000);
    float*     sbuf = (float*)((char*)bufA + 40960000);
    // bufB sub: x | x_f16
    float*     xnode = bufB;
    _Float16*  xf16  = (_Float16*)((char*)bufB + 20480000);

    // ============================= CNN =============================
    conv2d_tanh<<<cdivll(200LL*64*1024, TPB), TPB, 0, stream>>>(x_low, cw1, cb1, bufA, 200, 5, 64, 5, 2);
    conv2d_tanh<<<cdivll(200LL*32*1024, TPB), TPB, 0, stream>>>(bufA, cw2, cb2, bufB, 200, 64, 32, 3, 1);
    hipMemsetAsync(hbuf, 0, (size_t)NLOW * 128 * sizeof(_Float16), stream);
    conv3_shuffle_sig<<<cdivll(200LL*45*1024, TPB), TPB, 0, stream>>>(bufB, cw3, cb3, hbuf, 200);

    // ================= downscale GATv2 (H=1, C=25) =================
    // xl = h[73728,125(pad128)] @ Wl^T  -> [73728, 32(log 25)]   (WMMA)
    pack_weight_f16<<<cdivll(128*32, TPB), TPB, 0, stream>>>(dWl, wpack, 125, 25, 128, 32);
    launch_gemm(128, hbuf, wpack, 32, dbl, 25, xl_low, 32, 32, NLOW, 0, stream);
    xr_down<<<cdivll((long long)NH*25, TPB), TPB, 0, stream>>>(high_x, dWr, dbr, xr_high, NH, 25);

    hipMemsetAsync(mkey, 0, (size_t)NH * 4, stream);
    hipMemsetAsync(den,  0, (size_t)NH * 4, stream);
    hipMemsetAsync(cnt,  0, (size_t)NH * 4, stream);
    hipMemsetAsync(sbuf, 0, (size_t)NH * 25 * 4, stream);
    gat_logits<<<cdivll(E12, TPB), TPB, 0, stream>>>(xl_low, 32, xr_high, 25, datt, 1, 25,
                                                     src_l2h, dst_l2h, E12, 0, logits, mkey);
    gat_exp<<<cdivll(E12, TPB), TPB, 0, stream>>>(logits, dst_l2h, E12, 0, 1, mkey, exbuf, den, cnt);
    gat_scatter<<<cdivll((long long)E12*25, TPB), TPB, 0, stream>>>(xl_low, 32, exbuf, den,
                                                                   src_l2h, dst_l2h, E12, 0, 1, 25, sbuf);
    gat_finalize<<<cdivll((long long)NH*25, TPB), TPB, 0, stream>>>(sbuf, cnt, dbias, 25,
                                                                   xnode, 26, 1, NH, 0);
    copy_col0<<<cdivll(NH, TPB), TPB, 0, stream>>>(z_std, xnode, 26, NH);

    // ==================== processor: 5 GATv2 layers ====================
    const int Dl[5]   = {26, 128, 128, 128, 128};
    const int Dpad[5] = {32, 128, 128, 128, 128};
    const int HCl[5]  = {128, 128, 128, 128, 64};
    const int Hl[5]   = {2, 2, 2, 2, 1};
    for (int l = 0; l < 5; ++l) {
        const float* Wl   = (const float*)d_in[29 + 6*l + 0];
        const float* Wr   = (const float*)d_in[29 + 6*l + 1];
        const float* att  = (const float*)d_in[29 + 6*l + 2];
        const float* bias = (const float*)d_in[29 + 6*l + 3];
        const float* bl   = (const float*)d_in[29 + 6*l + 4];
        const float* br   = (const float*)d_in[29 + 6*l + 5];
        int D = Dl[l], Dp = Dpad[l], HC = HCl[l], H = Hl[l];

        bn_stats<<<D, TPB, 0, stream>>>(xnode, D, NH, bn_g[l], bn_b[l], bnscale, bnshift);
        bn_apply_f16<<<cdivll((long long)NH*Dp, TPB), TPB, 0, stream>>>(xnode, D, D, bnscale, bnshift,
                                                                        xf16, Dp, NH, l > 0);
        // xl = bn(x) @ Wl^T + bl   (WMMA)
        pack_weight_f16<<<cdivll(Dp*HC, TPB), TPB, 0, stream>>>(Wl, wpack, D, HC, Dp, HC);
        launch_gemm(Dp, xf16, wpack, HC, bl, HC, xlN, HC, HC, NH, 0, stream);
        // xr = bn(x) @ Wr^T + br   (WMMA)
        pack_weight_f16<<<cdivll(Dp*HC, TPB), TPB, 0, stream>>>(Wr, wpack, D, HC, Dp, HC);
        launch_gemm(Dp, xf16, wpack, HC, br, HC, xrN, HC, HC, NH, 0, stream);

        hipMemsetAsync(mkey, 0, (size_t)NH * H * 4, stream);
        hipMemsetAsync(den,  0, (size_t)NH * H * 4, stream);
        hipMemsetAsync(cnt,  0, (size_t)NH * 4, stream);
        hipMemsetAsync(sbuf, 0, (size_t)NH * HC * 4, stream);
        long long EH = (long long)(EHH + NH) * H;
        gat_logits<<<cdivll(EH, TPB), TPB, 0, stream>>>(xlN, HC, xrN, HC, att, H, 64,
                                                        src_hh, dst_hh, EHH, NH, logits, mkey);
        gat_exp<<<cdivll(EH, TPB), TPB, 0, stream>>>(logits, dst_hh, EHH, NH, H, mkey, exbuf, den, cnt);
        long long EHC = (long long)(EHH + NH) * HC;
        gat_scatter<<<cdivll(EHC, TPB), TPB, 0, stream>>>(xlN, HC, exbuf, den,
                                                          src_hh, dst_hh, EHH, NH, H, 64, sbuf);
        gat_finalize<<<cdivll((long long)NH*HC, TPB), TPB, 0, stream>>>(sbuf, cnt, bias, HC,
                                                                        xnode, HC, 0, NH, (l == 4));
    }

    // ======================== predictor MLP (WMMA) ========================
    // y1 = relu(x @ W1^T + b1), y2 = relu(y1 @ W2^T + b2), out = y2 @ W3^T + b3
    float*    y1    = xlN;                       // reuse (free)
    _Float16* y1f16 = (_Float16*)xrN;            // reuse (free)
    float*    y2    = sbuf;                      // reuse (free)
    _Float16* y2f16 = xf16;                      // reuse (free after GEMM1)

    cast_f16_pad<<<cdivll((long long)NH*64, TPB), TPB, 0, stream>>>(xnode, 64, 64, xf16, 64, NH);
    pack_weight_f16<<<cdivll(64*64, TPB), TPB, 0, stream>>>(pW1, wpack, 64, 64, 64, 64);
    launch_gemm(64, xf16, wpack, 64, pb1, 64, y1, 64, 64, NH, 1, stream);
    cast_f16_pad<<<cdivll((long long)NH*64, TPB), TPB, 0, stream>>>(y1, 64, 64, y1f16, 64, NH);
    pack_weight_f16<<<cdivll(64*32, TPB), TPB, 0, stream>>>(pW2, wpack, 64, 32, 64, 32);
    launch_gemm(64, y1f16, wpack, 32, pb2, 32, y2, 32, 32, NH, 1, stream);
    cast_f16_pad<<<cdivll((long long)NH*32, TPB), TPB, 0, stream>>>(y2, 32, 32, y2f16, 32, NH);
    pack_weight_f16<<<cdivll(32*16, TPB), TPB, 0, stream>>>(pW3, wpack, 32, 1, 32, 16);
    launch_gemm(32, y2f16, wpack, 16, pb3, 1, (float*)d_out, 1, 1, NH, 0, stream);
}
